// SimilarityLoss_75917841924187
// MI455X (gfx1250) — matrix operand order repaired and checked by reading, not compiled, not run on hardware
//
#include <hip/hip_runtime.h>

// ---------------------------------------------------------------------------
// SimilarityLoss for MI455X (gfx1250, wave32, WMMA bf16 16x16x32, f32 accum)
// One-time bf16 conversion (row-major + transposed v) in workspace so both
// GEMM phases run as pure clause-grouped global b128 loads + v_wmma.
// ---------------------------------------------------------------------------

typedef __attribute__((ext_vector_type(16))) __bf16 v16bf;
typedef __attribute__((ext_vector_type(8)))  __bf16 v8bf;
typedef __attribute__((ext_vector_type(8)))  float  v8f;

namespace sl {
constexpr int N = 16, T = 64, D = 1024, V = 1024, M = 16, HW = 64;
constexpr float G1 = 5.0f, INV_G2 = 0.2f, G3 = 10.0f;

// LDS layout (bytes). Row pads chosen so bf16 rows are 16B aligned
// (ds_load_b128) and shift 4 banks per row (conflict-free b128 frag loads).
constexpr int ATT_STRIDE = 1032;  // bf16/row: 2064B, %16==0, bank shift 4
constexpr int SIM_STRIDE = 65;    // f32/row : bank shift 1
constexpr int VTS_STRIDE = 136;   // bf16/row: 272B, %16==0, bank shift 4

constexpr int OFF_ATT  = 0;
constexpr int OFF_SIM  = OFF_ATT  + T * ATT_STRIDE * 2;   // 132096
constexpr int OFF_VTS  = OFF_SIM  + T * SIM_STRIDE * 4;   // +16640
constexpr int OFF_CMAX = OFF_VTS  + 64 * VTS_STRIDE * 2;  // +17408
constexpr int OFF_CSUM = OFF_CMAX + 64 * 4;
constexpr int OFF_BETA = OFF_CSUM + 64 * 4;
constexpr int OFF_VN2  = OFF_BETA + 64 * 4;
constexpr int OFF_VDE  = OFF_VN2  + 64 * 4;
constexpr int OFF_RED  = OFF_VDE  + 64 * 4;
constexpr int OFF_SC   = OFF_RED  + 256 * 4;
constexpr int LDS_BYTES = OFF_SC + 32;                    // = 168480

// workspace float counts
constexpr size_t WS_HDR_FLOATS = (size_t)N * N + (size_t)N * T * M +
                                 (size_t)N * V + (size_t)N * T;  // 34048
constexpr size_t E16_ELEMS = (size_t)N * T * D;                  // 1M
constexpr size_t V16_ELEMS = (size_t)N * V * D;                  // 16M
}  // namespace sl

__device__ __forceinline__ void cvt4(v16bf& d, int base, float4 f) {
  d[base + 0] = (__bf16)f.x; d[base + 1] = (__bf16)f.y;
  d[base + 2] = (__bf16)f.z; d[base + 3] = (__bf16)f.w;
}

__device__ __forceinline__ v16bf ld16(const __bf16* p0, const __bf16* p1) {
  v8bf x = *(const v8bf*)p0;
  v8bf y = *(const v8bf*)p1;
  v16bf r;
#pragma unroll
  for (int i = 0; i < 8; ++i) { r[i] = x[i]; r[8 + i] = y[i]; }
  return r;
}

// ---------------- norm precompute (+ optional bf16 conversion) -------------
__global__ void sl_norms_kernel(const float* __restrict__ vB,
                                const float* __restrict__ eB,
                                float* __restrict__ vnormG,
                                float* __restrict__ enormG,
                                __bf16* __restrict__ v16B,
                                __bf16* __restrict__ e16B, int doCvt) {
  using namespace sl;
  int id = blockIdx.x * blockDim.x + threadIdx.x;
  const float* p;
  __bf16* q = nullptr;
  float* outp;
  if (id < N * V) {
    p = vB + (size_t)id * D;
    if (doCvt) q = v16B + (size_t)id * D;
    outp = vnormG + id;
  } else if (id < N * V + N * T) {
    int j = id - N * V;
    p = eB + (size_t)j * D;
    if (doCvt) q = e16B + (size_t)j * D;
    outp = enormG + j;
  } else {
    return;
  }
  float s = 0.f;
  for (int d = 0; d < D; d += 8) {
    float4 f0 = *(const float4*)(p + d);
    float4 f1 = *(const float4*)(p + d + 4);
    s += f0.x * f0.x + f0.y * f0.y + f0.z * f0.z + f0.w * f0.w;
    s += f1.x * f1.x + f1.y * f1.y + f1.z * f1.z + f1.w * f1.w;
    if (doCvt) {
      v8bf o;
      o[0] = (__bf16)f0.x; o[1] = (__bf16)f0.y;
      o[2] = (__bf16)f0.z; o[3] = (__bf16)f0.w;
      o[4] = (__bf16)f1.x; o[5] = (__bf16)f1.y;
      o[6] = (__bf16)f1.z; o[7] = (__bf16)f1.w;
      *(v8bf*)(q + d) = o;
    }
  }
  *outp = sqrtf(s);
}

// ---------------- tiled transpose: v16T[a][d][r] = bf16(v[a][r][d]) --------
__global__ __launch_bounds__(256) void sl_transpose_kernel(
    const float* __restrict__ vB, __bf16* __restrict__ v16T) {
  using namespace sl;
  __shared__ __bf16 tile[64 * 72];
  const int tid = threadIdx.x;
  const int a  = blockIdx.x >> 8;
  const int tt = blockIdx.x & 255;
  const int r0 = (tt >> 4) * 64;
  const int d0 = (tt & 15) * 64;
  // coalesced f32 reads: tile[i][j] = v[a][r0+i][d0+j]
  for (int li = tid; li < 64 * 16; li += 256) {
    int i = li >> 4, j4 = (li & 15) * 4;
    float4 f = *(const float4*)(vB + ((size_t)a * V + r0 + i) * D + d0 + j4);
    tile[i * 72 + j4 + 0] = (__bf16)f.x;
    tile[i * 72 + j4 + 1] = (__bf16)f.y;
    tile[i * 72 + j4 + 2] = (__bf16)f.z;
    tile[i * 72 + j4 + 3] = (__bf16)f.w;
  }
  __syncthreads();
  // coalesced bf16 writes: v16T[a][d0+i][r0+c] = tile[c][i]
  for (int li = tid; li < 64 * 8; li += 256) {
    int i = li >> 3, c8 = (li & 7) * 8;
    v8bf o;
#pragma unroll
    for (int k = 0; k < 8; ++k) o[k] = tile[(c8 + k) * 72 + i];
    *(v8bf*)(v16T + ((size_t)a * D + d0 + i) * V + r0 + c8) = o;
  }
}

// --------------------------- per-pair kernel -------------------------------
// grid (b, a): blockIdx.x = e index b, blockIdx.y = v index a. 256 threads.
// PRE: bf16 copies of e/v available. TR: transposed bf16 v available
// (phase 2 B-fragments load straight from global, no LDS staging).
template <bool PRE, bool TR>
__global__ __launch_bounds__(256) void sl_pair_kernel(
    const float* __restrict__ eB, const float* __restrict__ vB,
    const __bf16* __restrict__ e16B, const __bf16* __restrict__ v16B,
    const __bf16* __restrict__ v16TB,
    const float* __restrict__ vnormG, const float* __restrict__ enormG,
    float* __restrict__ Sg, float* __restrict__ betasG) {
  using namespace sl;
  extern __shared__ char smem[];
  __bf16* att  = (__bf16*)(smem + OFF_ATT);   // exp(G1*nsim), bf16, T x V
  float*  sim  = (float*)(smem + OFF_SIM);    // 64x64 tile scratch
  __bf16* vts  = (__bf16*)(smem + OFF_VTS);   // v^T stage (non-TR path)
  float*  cmax = (float*)(smem + OFF_CMAX);
  float*  csum = (float*)(smem + OFF_CSUM);   // reciprocal of column sums
  float*  beta = (float*)(smem + OFF_BETA);
  float*  vn2  = (float*)(smem + OFF_VN2);
  float*  vde  = (float*)(smem + OFF_VDE);
  float*  red  = (float*)(smem + OFF_RED);
  float*  sc   = (float*)(smem + OFF_SC);

  const int b = blockIdx.x;
  const int a = blockIdx.y;
  const float*  e    = eB + (size_t)b * T * D;
  const float*  v    = vB + (size_t)a * V * D;
  const __bf16* e16  = e16B + (size_t)b * T * D;
  const __bf16* v16  = v16B + (size_t)a * V * D;
  const __bf16* v16T = v16TB + (size_t)a * D * V;

  const int tid  = threadIdx.x;
  const int wv   = tid >> 5, lane = tid & 31;
  const int lm   = lane & 15, lh = lane >> 4;
  const int wrow = (wv & 3) * 16;   // output tile row base
  const int wcol = (wv >> 2) * 32;  // output tile col base (2 subtiles)

  if (tid == 0) sc[2] = 0.0f;               // sum_m exp(ref_m)
  if (tid < T) { vn2[tid] = 0.0f; vde[tid] = 0.0f; }
  __syncthreads();

  // ================= Phase 1: sim = e @ v^T, per 64-region block ==========
  for (int m = 0; m < M; ++m) {
    v8f acc0 = {}; v8f acc1 = {};
    if constexpr (PRE) {
      const __bf16* ea  = e16 + (size_t)(wrow + lm) * D;
      const __bf16* vb0 = v16 + (size_t)(m * HW + wcol + lm) * D;
      const __bf16* vb1 = v16 + (size_t)(m * HW + wcol + 16 + lm) * D;
#pragma unroll 2
      for (int d0 = 0; d0 < D; d0 += 32) {
        // A (16x32): lane holds row m, K = lh*16 + 0..15 contiguous
        v16bf af  = ld16(ea + d0 + lh * 16, ea + d0 + lh * 16 + 8);
        // B (32x16): lane holds column n, K = lh*16 + 0..15 contiguous
        v16bf bf0 = ld16(vb0 + d0 + lh * 16, vb0 + d0 + lh * 16 + 8);
        v16bf bf1 = ld16(vb1 + d0 + lh * 16, vb1 + d0 + lh * 16 + 8);
        acc0 = __builtin_amdgcn_wmma_f32_16x16x32_bf16(false, af, false, bf0,
                                                       (short)0, acc0, false, false);
        acc1 = __builtin_amdgcn_wmma_f32_16x16x32_bf16(false, af, false, bf1,
                                                       (short)0, acc1, false, false);
      }
    } else {
      const float* ea  = e + (size_t)(wrow + lm) * D;
      const float* vb0 = v + (size_t)(m * HW + wcol + lm) * D;
      const float* vb1 = v + (size_t)(m * HW + wcol + 16 + lm) * D;
      for (int d0 = 0; d0 < D; d0 += 32) {
        v16bf af, bf0, bf1;
        cvt4(af, 0,  *(const float4*)(ea + d0 + lh * 16));
        cvt4(af, 4,  *(const float4*)(ea + d0 + lh * 16 + 4));
        cvt4(af, 8,  *(const float4*)(ea + d0 + lh * 16 + 8));
        cvt4(af, 12, *(const float4*)(ea + d0 + lh * 16 + 12));
        cvt4(bf0, 0,  *(const float4*)(vb0 + d0 + lh * 16));
        cvt4(bf0, 4,  *(const float4*)(vb0 + d0 + lh * 16 + 4));
        cvt4(bf0, 8,  *(const float4*)(vb0 + d0 + lh * 16 + 8));
        cvt4(bf0, 12, *(const float4*)(vb0 + d0 + lh * 16 + 12));
        cvt4(bf1, 0,  *(const float4*)(vb1 + d0 + lh * 16));
        cvt4(bf1, 4,  *(const float4*)(vb1 + d0 + lh * 16 + 4));
        cvt4(bf1, 8,  *(const float4*)(vb1 + d0 + lh * 16 + 8));
        cvt4(bf1, 12, *(const float4*)(vb1 + d0 + lh * 16 + 12));
        acc0 = __builtin_amdgcn_wmma_f32_16x16x32_bf16(false, af, false, bf0,
                                                       (short)0, acc0, false, false);
        acc1 = __builtin_amdgcn_wmma_f32_16x16x32_bf16(false, af, false, bf1,
                                                       (short)0, acc1, false, false);
      }
    }
    // dump tile: C layout VGPR i, lanes 0-15 -> M=i, lanes 16-31 -> M=i+8
#pragma unroll
    for (int i = 0; i < 8; ++i) {
      sim[(wrow + i + 8 * lh) * SIM_STRIDE + wcol + lm]      = acc0[i];
      sim[(wrow + i + 8 * lh) * SIM_STRIDE + wcol + 16 + lm] = acc1[i];
    }
    __syncthreads();

    // column softmax stats over words t (axis=0)
    if (tid < HW) {
      int j = tid;
      float mx = -3.4e38f;
      for (int t = 0; t < T; ++t) mx = fmaxf(mx, sim[t * SIM_STRIDE + j]);
      float s = 0.f;
      for (int t = 0; t < T; ++t) s += __expf(sim[t * SIM_STRIDE + j] - mx);
      cmax[j] = mx;
      csum[j] = 1.0f / s;
    }
    __syncthreads();

    // per row: se_sum (-> beta), attn numerators exp(G1*nsim) (bf16 -> LDS)
    if (tid < T) {
      int t = tid;
      float rs = 0.f;
      __bf16* arow = att + t * ATT_STRIDE + m * HW;
      for (int j = 0; j < HW; ++j) {
        float ns = __expf(sim[t * SIM_STRIDE + j] - cmax[j]) * csum[j];
        rs += __expf(ns);
        arow[j] = (__bf16)__expf(G1 * ns);  // row-scale cancels in cos1
      }
      red[t] = rs;
    }
    __syncthreads();
    if (tid == 0) {
      float bs = 0.f;
      for (int t = 0; t < T; ++t) bs += red[t];
      sc[0] = 1.0f / bs;
    }
    __syncthreads();
    if (tid < T) {
      float bt = red[tid] * sc[0];
      beta[tid] = bt;
      if (a == b) betasG[((size_t)a * T + tid) * M + m] = bt;
    }
    __syncthreads();

    // ||e_prime[:,m]|| = || e^T beta ||   (f32 inputs for accuracy)
    float part = 0.f;
    for (int d = tid; d < D; d += 256) {
      float s = 0.f;
      for (int t = 0; t < T; ++t) s += beta[t] * e[t * D + d];
      part += s * s;
    }
    red[tid] = part;
    __syncthreads();
    if (tid == 0) {
      float s = 0.f;
      for (int i = 0; i < 256; ++i) s += red[i];
      sc[1] = 1.0f / sqrtf(s);
    }
    __syncthreads();

    // cos2[j] = (beta . sim[:,j]) / (||v_r|| * ||e_prime||) ; ref = mean_j
    if (tid < HW) {
      int j = tid;
      float dot = 0.f;
      for (int t = 0; t < T; ++t) dot += beta[t] * sim[t * SIM_STRIDE + j];
      red[j] = dot * sc[1] / vnormG[a * V + m * HW + j];
    }
    __syncthreads();
    if (tid == 0) {
      float s = 0.f;
      for (int j = 0; j < HW; ++j) s += red[j];
      sc[2] += __expf(s * (1.0f / HW));
    }
    __syncthreads();
  }

  // ====== Phase 2: v_tilde = attn @ v, per 64-col d-block (K = 1024) ======
  for (int db = 0; db < 16; ++db) {
    v8f acc0 = {}; v8f acc1 = {};
    if constexpr (TR) {
      // B[k][n] = v[k][db*64+n] = v16T[db*64+n][k]: contiguous global loads.
      const __bf16* ar  = att + (wrow + lm) * ATT_STRIDE;
      const __bf16* vt0 = v16T + (size_t)(db * 64 + wcol + lm) * V;
      const __bf16* vt1 = v16T + (size_t)(db * 64 + wcol + 16 + lm) * V;
#pragma unroll 2
      for (int kg = 0; kg < V; kg += 32) {
        v16bf af  = ld16(ar + kg + lh * 16, ar + kg + lh * 16 + 8);
        v16bf bf0 = ld16(vt0 + kg + lh * 16, vt0 + kg + lh * 16 + 8);
        v16bf bf1 = ld16(vt1 + kg + lh * 16, vt1 + kg + lh * 16 + 8);
        acc0 = __builtin_amdgcn_wmma_f32_16x16x32_bf16(false, af, false, bf0,
                                                       (short)0, acc0, false, false);
        acc1 = __builtin_amdgcn_wmma_f32_16x16x32_bf16(false, af, false, bf1,
                                                       (short)0, acc1, false, false);
      }
    } else {
      for (int ks = 0; ks < 8; ++ks) {         // stage 128 region rows in LDS
        __syncthreads();
        if constexpr (PRE) {
          for (int li = tid; li < 128 * 8; li += 256) {
            int kk = li >> 3, n8 = (li & 7) * 8;
            v8bf f = *(const v8bf*)(v16 + (size_t)(ks * 128 + kk) * D + db * 64 + n8);
#pragma unroll
            for (int c = 0; c < 8; ++c) vts[(n8 + c) * VTS_STRIDE + kk] = f[c];
          }
        } else {
          for (int li = tid; li < 128 * 16; li += 256) {
            int kk = li >> 4, n4 = (li & 15) * 4;
            float4 f = *(const float4*)(v + (size_t)(ks * 128 + kk) * D + db * 64 + n4);
            vts[(n4 + 0) * VTS_STRIDE + kk] = (__bf16)f.x;
            vts[(n4 + 1) * VTS_STRIDE + kk] = (__bf16)f.y;
            vts[(n4 + 2) * VTS_STRIDE + kk] = (__bf16)f.z;
            vts[(n4 + 3) * VTS_STRIDE + kk] = (__bf16)f.w;
          }
        }
        __syncthreads();
        for (int kk0 = 0; kk0 < 128; kk0 += 32) {
          int kg = ks * 128 + kk0;
          const __bf16* ar = att + (wrow + lm) * ATT_STRIDE + kg;
          v16bf af = ld16(ar + lh * 16, ar + lh * 16 + 8);
          const __bf16* b0 = vts + (wcol + lm) * VTS_STRIDE + kk0 + lh * 16;
          const __bf16* b1 = vts + (wcol + 16 + lm) * VTS_STRIDE + kk0 + lh * 16;
          v16bf bf0 = ld16(b0, b0 + 8);
          v16bf bf1 = ld16(b1, b1 + 8);
          acc0 = __builtin_amdgcn_wmma_f32_16x16x32_bf16(false, af, false, bf0,
                                                         (short)0, acc0, false, false);
          acc1 = __builtin_amdgcn_wmma_f32_16x16x32_bf16(false, af, false, bf1,
                                                         (short)0, acc1, false, false);
        }
      }
    }
    __syncthreads();
#pragma unroll
    for (int i = 0; i < 8; ++i) {
      sim[(wrow + i + 8 * lh) * SIM_STRIDE + wcol + lm]      = acc0[i];
      sim[(wrow + i + 8 * lh) * SIM_STRIDE + wcol + 16 + lm] = acc1[i];
    }
    __syncthreads();
    if (tid < T) {
      int t = tid;
      const float* er = e + t * D + db * 64;
      float sn = 0.f, sd = 0.f;
      for (int j = 0; j < 64; ++j) {
        float x = sim[t * SIM_STRIDE + j];
        sn += x * x;
        sd += x * er[j];
      }
      vn2[t] += sn;
      vde[t] += sd;
    }
  }
  __syncthreads();

  if (tid == 0) {
    float c1 = 0.f;
    for (int t = 0; t < T; ++t)
      c1 += vde[t] / (sqrtf(vn2[t]) * enormG[b * T + t]);
    Sg[a * N + b] = INV_G2 * logf(c1) + INV_G2 * logf(sc[2]);
  }
}

// --------------------------- finalize --------------------------------------
__global__ __launch_bounds__(64) void sl_finalize_kernel(
    const float* __restrict__ Sg, const float* __restrict__ betasG,
    float* __restrict__ out) {
  using namespace sl;
  __shared__ float es[N * N];
  __shared__ float diag[T];
  __shared__ float red[64];
  const int tid = threadIdx.x;

  for (int i = tid; i < N * N; i += 64) es[i] = G3 * __expf(Sg[i]);
  __syncthreads();

  float l12 = 0.f;
  if (tid == 0) {
    float l1 = 0.f, l2 = 0.f;
    for (int i = 0; i < N; ++i) {
      float rs = 0.f, cs = 0.f;
      for (int j = 0; j < N; ++j) { rs += es[i * N + j]; cs += es[j * N + i]; }
      l1 += es[i * N + i] / rs;   // P_DQ (sum over e, axis=1)
      l2 += es[i * N + i] / cs;   // P_QD (sum over v, axis=0)
    }
    l12 = -(l1 + l2);
  }

  float totreg = 0.f;
  for (int i = 0; i < N; ++i) {
    const float* bt = betasG + (size_t)i * T * M;
    {
      float dsum = 0.f;
      for (int m = 0; m < M; ++m) { float x = bt[tid * M + m]; dsum += x * x; }
      diag[tid] = dsum;           // B[s,s]
    }
    __syncthreads();
    float sq = 0.f;
    for (int s = 0; s < T; ++s) {
      float dot = 0.f;
      for (int m = 0; m < M; ++m) dot += bt[tid * M + m] * bt[s * M + m];
      float val = dot - diag[s];  // B[t,s] - diag broadcast across rows
      sq += val * val;
    }
    red[tid] = sq;
    __syncthreads();
    if (tid == 0) {
      float s = 0.f;
      for (int k = 0; k < 64; ++k) s += red[k];
      totreg += sqrtf(s);
    }
    __syncthreads();
  }
  if (tid == 0) out[0] = l12 + totreg;
}

// --------------------------- launch ----------------------------------------
extern "C" void kernel_launch(void* const* d_in, const int* in_sizes, int n_in,
                              void* d_out, int out_size, void* d_ws, size_t ws_size,
                              hipStream_t stream) {
  using namespace sl;
  (void)in_sizes; (void)n_in; (void)out_size;

  const float* eB = (const float*)d_in[0];  // (16, 64, 1024) f32
  const float* vB = (const float*)d_in[1];  // (16, 1024, 1024) f32

  float* ws    = (float*)d_ws;
  float* S     = ws;                        // 256
  float* betas = S + N * N;                 // 16*64*16 = 16384
  float* vnorm = betas + N * T * M;         // 16*1024  = 16384
  float* enorm = vnorm + N * V;             // 16*64    = 1024
  __bf16* e16  = (__bf16*)(ws + WS_HDR_FLOATS);          // 16B-aligned
  __bf16* v16  = e16 + E16_ELEMS;
  __bf16* v16T = v16 + V16_ELEMS;

  const size_t need_pre = WS_HDR_FLOATS * 4 + (E16_ELEMS + V16_ELEMS) * 2;
  const size_t need_tr  = need_pre + V16_ELEMS * 2;
  const bool pre = ws_size >= need_pre;
  const bool tr  = ws_size >= need_tr;

  (void)hipFuncSetAttribute((const void*)sl_pair_kernel<true, true>,
                            hipFuncAttributeMaxDynamicSharedMemorySize, LDS_BYTES);
  (void)hipFuncSetAttribute((const void*)sl_pair_kernel<true, false>,
                            hipFuncAttributeMaxDynamicSharedMemorySize, LDS_BYTES);
  (void)hipFuncSetAttribute((const void*)sl_pair_kernel<false, false>,
                            hipFuncAttributeMaxDynamicSharedMemorySize, LDS_BYTES);

  int nrows = N * V + N * T;
  sl_norms_kernel<<<(nrows + 255) / 256, 256, 0, stream>>>(
      vB, eB, vnorm, enorm, v16, e16, pre ? 1 : 0);
  if (tr) {
    sl_transpose_kernel<<<N * 256, 256, 0, stream>>>(vB, v16T);
    sl_pair_kernel<true, true><<<dim3(N, N), 256, LDS_BYTES, stream>>>(
        eB, vB, e16, v16, v16T, vnorm, enorm, S, betas);
  } else if (pre) {
    sl_pair_kernel<true, false><<<dim3(N, N), 256, LDS_BYTES, stream>>>(
        eB, vB, e16, v16, v16T, vnorm, enorm, S, betas);
  } else {
    sl_pair_kernel<false, false><<<dim3(N, N), 256, LDS_BYTES, stream>>>(
        eB, vB, e16, v16, v16T, vnorm, enorm, S, betas);
  }
  sl_finalize_kernel<<<1, 64, 0, stream>>>(S, betas, (float*)d_out);
}
